// Encoding_34514357191359
// MI455X (gfx1250) — compile-verified, matrix operand order checked by reading, not run on hardware
//
#include <hip/hip_runtime.h>
#include <stdint.h>

typedef __attribute__((ext_vector_type(16))) __bf16 bf16x16;
typedef __attribute__((ext_vector_type(8)))  float  f32x8;

#define BATCH 16
#define CH    512
#define NIMG  9216
#define KCW   32
#define NT    128
#define TILES_PER_WG 4
#define WG_PER_BATCH 18          // 9216 / (128*4)
#define NWG   (BATCH*WG_PER_BATCH)

// padded LDS strides (elements) chosen for bank spread + 8B alignment
#define WT_S   516
#define XCN_S  132
#define XNC_S  516
#define AWT_S  132
#define XC_S   33

#define OFF_WT   0
#define OFF_XCN  (OFF_WT  + KCW*WT_S*2)       //  33024
#define OFF_XNC  (OFF_XCN + CH*XCN_S*2)       // 168192
#define OFF_AWT  OFF_XNC                      // aw^T aliases dead n-major tile
#define OFF_XC   (OFF_XNC + NT*XNC_S*2)       // 300288
#define OFF_X2P  (OFF_XC  + NT*XC_S*4)        // 317184
#define OFF_C2   (OFF_X2P + 8*NT*4)           // 321280
#define OFF_SC   (OFF_C2  + KCW*4)            // 321408
#define SMEM_BYTES (OFF_SC + KCW*4)           // 321536 B = 314 KB (< 320 KB/WGP)

__device__ __forceinline__ uint16_t f2bf(float f) {
  union { float f; uint32_t u; } v; v.f = f;
  uint32_t r = v.u + 0x7FFFu + ((v.u >> 16) & 1u);   // round-to-nearest-even
  return (uint16_t)(r >> 16);
}
__device__ __forceinline__ float bf2f(uint16_t h) {
  union { uint32_t u; float f; } v; v.u = ((uint32_t)h) << 16;
  return v.f;
}

// Load one 16x32 bf16 A/B WMMA fragment from LDS (row-major source, 16-bit layout
// per CDNA5 ISA: lane<16 -> K 0..7 & 16..23, lane>=16 -> K 8..15 & 24..31).
__device__ __forceinline__ bf16x16 load_frag(const uint16_t* base, int stride,
                                             int row, int c0, int lh) {
  union { bf16x16 v; uint2 u[4]; } f;
  const uint16_t* p = base + row * stride + c0 + lh * 8;
  f.u[0] = *(const uint2*)(p +  0);
  f.u[1] = *(const uint2*)(p +  4);
  f.u[2] = *(const uint2*)(p + 16);
  f.u[3] = *(const uint2*)(p + 20);
  return f.v;
}

__device__ __forceinline__ f32x8 wmma_bf16(bf16x16 a, bf16x16 b, f32x8 c) {
  return __builtin_amdgcn_wmma_f32_16x16x32_bf16(false, a, false, b, (short)0, c,
                                                 false, false);
}

__global__ __launch_bounds__(256) void enc_main(const float* __restrict__ x,
                                                const float* __restrict__ cw,
                                                const float* __restrict__ scale,
                                                float* __restrict__ ws_enc,
                                                float* __restrict__ ws_awsum) {
  __shared__ __align__(16) unsigned char smem[SMEM_BYTES];
  uint16_t* wt   = (uint16_t*)(smem + OFF_WT);   // codewords bf16 [32][516]
  uint16_t* xcn  = (uint16_t*)(smem + OFF_XCN);  // x tile c-major [512][132]
  uint16_t* xnc  = (uint16_t*)(smem + OFF_XNC);  // x tile n-major [128][516]
  uint16_t* awt  = (uint16_t*)(smem + OFF_AWT);  // aw^T bf16 [32][132] (alias)
  float*    xcf  = (float*)(smem + OFF_XC);      // xc f32 [128][33]
  float*    x2pf = (float*)(smem + OFF_X2P);     // x^2 partials [8][128]
  float*    c2f  = (float*)(smem + OFF_C2);
  float*    scf  = (float*)(smem + OFF_SC);

  const int tid  = threadIdx.x;
  const int wid  = tid >> 5;
  const int lane = tid & 31;
  const int lm   = lane & 15;
  const int lh   = lane >> 4;

  const int b     = blockIdx.x / WG_PER_BATCH;
  const int pslot = blockIdx.x % WG_PER_BATCH;
  const float* xb = x + (size_t)b * CH * NIMG;

  // ---- one-time: codewords -> bf16 LDS, c2 = ||c_k||^2 (exact f32), scale
  for (int i = tid; i < KCW * CH; i += 256) {
    int k = i >> 9, c = i & (CH - 1);
    wt[k * WT_S + c] = f2bf(cw[i]);
  }
  if (tid < KCW) {
    const float* r = cw + tid * CH;
    float s = 0.f;
    for (int c = 0; c < CH; ++c) { float v = r[c]; s += v * v; }
    c2f[tid] = s;
    scf[tid] = scale[tid];
  }
  __syncthreads();

  // persistent enc accumulators: wave -> (kt = wid&1, ct = (wid>>1)+4j), j=0..7
  f32x8 bacc[8];
  for (int j = 0; j < 8; ++j)
    for (int e = 0; e < 8; ++e) bacc[j][e] = 0.f;
  float awsum_reg = 0.f;
  const int kt  = wid & 1;
  const int ctb = wid >> 1;

  for (int it = 0; it < TILES_PER_WG; ++it) {
    const int n0g = (pslot * TILES_PER_WG + it) * NT;

    // ---- Phase 0: stream x tile (f32, coalesced) -> bf16 c-major LDS + f32 x^2
    {
      int nsub = tid & 31, cph = tid >> 5;   // wave reads 512B contiguous per row
      float s0 = 0.f, s1 = 0.f, s2 = 0.f, s3 = 0.f;
      for (int i = 0; i < 64; ++i) {
        int c = cph + 8 * i;
        float4 v = *((const float4*)(xb + (size_t)c * NIMG + n0g) + nsub);
        s0 += v.x * v.x; s1 += v.y * v.y; s2 += v.z * v.z; s3 += v.w * v.w;
        uint2 pk;
        pk.x = (uint32_t)f2bf(v.x) | ((uint32_t)f2bf(v.y) << 16);
        pk.y = (uint32_t)f2bf(v.z) | ((uint32_t)f2bf(v.w) << 16);
        *(uint2*)&xcn[c * XCN_S + nsub * 4] = pk;
      }
      float* xp = x2pf + cph * NT + nsub * 4;
      xp[0] = s0; xp[1] = s1; xp[2] = s2; xp[3] = s3;
    }
    __syncthreads();

    // ---- Phase 0c: build n-major copy (pair-packed u32 stores, conflict-light)
    for (int jj = 0; jj < 16; ++jj) {
      int job = tid + 256 * jj;
      int cp = job & 255, no = job >> 8;
      int c = cp * 2, n0 = no * 8;
      union { uint2 u[2]; uint16_t s[8]; } ra, rb;
      ra.u[0] = *(const uint2*)&xcn[c * XCN_S + n0];
      ra.u[1] = *(const uint2*)&xcn[c * XCN_S + n0 + 4];
      rb.u[0] = *(const uint2*)&xcn[(c + 1) * XCN_S + n0];
      rb.u[1] = *(const uint2*)&xcn[(c + 1) * XCN_S + n0 + 4];
      for (int n = 0; n < 8; ++n) {
        uint32_t pk = (uint32_t)ra.s[n] | ((uint32_t)rb.s[n] << 16);
        *(uint32_t*)&xnc[(n0 + n) * XNC_S + c] = pk;
      }
    }
    __syncthreads();

    // ---- Phase A: xc[n][k] = x . w  (wave wid owns n-subtile wid, both k-tiles)
    {
      f32x8 a0, a1;
      for (int e = 0; e < 8; ++e) { a0[e] = 0.f; a1[e] = 0.f; }
#pragma unroll 4
      for (int cc = 0; cc < 16; ++cc) {
        int c0 = cc * 32;
        bf16x16 af = load_frag(xnc, XNC_S, wid * 16 + lm, c0, lh); // A: x rows n
        bf16x16 b0 = load_frag(wt, WT_S, lm, c0, lh);              // B: cw cols k
        bf16x16 b1 = load_frag(wt, WT_S, 16 + lm, c0, lh);
        a0 = wmma_bf16(af, b0, a0);
        a1 = wmma_bf16(af, b1, a1);
      }
#pragma unroll
      for (int r = 0; r < 8; ++r) {             // D: lane holds (M=r+8*lh, N=lm)
        int n = wid * 16 + r + 8 * lh;
        xcf[n * XC_S + lm]      = a0[r];
        xcf[n * XC_S + 16 + lm] = a1[r];
      }
    }
    __syncthreads();

    // ---- softmax over K=32 (f32), write aw^T bf16 over dead n-major tile
    if (tid < NT) {
      int n = tid;
      float x2 = 0.f;
      for (int p = 0; p < 8; ++p) x2 += x2pf[p * NT + n];
      float v[KCW], mx = -3.0e38f;
#pragma unroll
      for (int k = 0; k < KCW; ++k) {
        float t = scf[k] * (x2 - 2.f * xcf[n * XC_S + k] + c2f[k]);
        v[k] = t; mx = fmaxf(mx, t);
      }
      float s = 0.f;
#pragma unroll
      for (int k = 0; k < KCW; ++k) { float e = __expf(v[k] - mx); v[k] = e; s += e; }
      float inv = 1.f / s;
#pragma unroll
      for (int k = 0; k < KCW; ++k) awt[k * AWT_S + n] = f2bf(v[k] * inv);
    }
    __syncthreads();

    // ---- per-k sum of aw (deterministic, wave 0 only; uniform per wave)
    if (tid < KCW) {
      float s = 0.f;
      for (int n = 0; n < NT; ++n) s += bf2f(awt[tid * AWT_S + n]);
      awsum_reg += s;
    }

    // ---- Phase B: enc[k][c] += aw^T . x  (accumulate across all 4 tiles)
#pragma unroll
    for (int sstep = 0; sstep < 4; ++sstep) {
      int n0 = sstep * 32;
      bf16x16 afr = load_frag(awt, AWT_S, kt * 16 + lm, n0, lh);  // A: aw rows k
#pragma unroll
      for (int j = 0; j < 8; ++j) {
        int ct = ctb + 4 * j;
        bf16x16 bfr = load_frag(xcn, XCN_S, ct * 16 + lm, n0, lh); // B: x rows c
        bacc[j] = wmma_bf16(afr, bfr, bacc[j]);
      }
    }
    __syncthreads();   // protect LDS tiles before next iteration overwrites
  }

  // ---- write this WG's 32x512 f32 partial + aw row-sums (no atomics)
  float* part = ws_enc + ((size_t)blockIdx.x << 14);
#pragma unroll
  for (int j = 0; j < 8; ++j) {
    int ct = ctb + 4 * j;
#pragma unroll
    for (int r = 0; r < 8; ++r) {
      int k = kt * 16 + r + 8 * lh;
      int c = ct * 16 + lm;
      part[k * CH + c] = bacc[j][r];
    }
  }
  if (tid < KCW) ws_awsum[blockIdx.x * KCW + tid] = awsum_reg;
}

// Deterministic 18-way reduction + "- (sum_n aw) * codewords" correction.
__global__ __launch_bounds__(256) void enc_reduce(const float* __restrict__ ws_enc,
                                                  const float* __restrict__ ws_awsum,
                                                  const float* __restrict__ cw,
                                                  float* __restrict__ out) {
  int idx = blockIdx.x * 256 + threadIdx.x;   // 0 .. 262143
  int bb  = idx >> 14;                        // batch
  int rem = idx & 16383;                      // k*512 + c
  int k   = rem >> 9;
  float s = 0.f, a = 0.f;
  for (int p = 0; p < WG_PER_BATCH; ++p) {
    int slot = bb * WG_PER_BATCH + p;
    s += ws_enc[((size_t)slot << 14) + rem];
    a += ws_awsum[slot * KCW + k];
  }
  out[idx] = s - a * cw[rem];
}

extern "C" void kernel_launch(void* const* d_in, const int* in_sizes, int n_in,
                              void* d_out, int out_size, void* d_ws, size_t ws_size,
                              hipStream_t stream) {
  (void)in_sizes; (void)n_in; (void)out_size; (void)ws_size;
  const float* x  = (const float*)d_in[0];
  const float* cw = (const float*)d_in[1];
  const float* sc = (const float*)d_in[2];
  float* out      = (float*)d_out;
  float* ws_enc   = (float*)d_ws;                                   // 288*16384 f32
  float* ws_awsum = ws_enc + (size_t)NWG * KCW * CH;                // 288*32 f32

  enc_main<<<NWG, 256, 0, stream>>>(x, cw, sc, ws_enc, ws_awsum);
  enc_reduce<<<(BATCH * KCW * CH) / 256, 256, 0, stream>>>(ws_enc, ws_awsum, cw, out);
}
